// CausalSelfAttention_10642928959915
// MI455X (gfx1250) — matrix operand order repaired and checked by reading, not compiled, not run on hardware
//
#include <hip/hip_runtime.h>
#include <hip/hip_bf16.h>
#include <stdint.h>

typedef __bf16 bf16;
typedef __attribute__((ext_vector_type(16))) __bf16 bf16x16;
typedef __attribute__((ext_vector_type(8)))  __bf16 bf16x8;
typedef __attribute__((ext_vector_type(4)))  __bf16 bf16x4;
typedef __attribute__((ext_vector_type(8)))  float  f32x8;
typedef __attribute__((ext_vector_type(4)))  float  f32x4;

#define WMMA_BF16(A, B, C) \
  __builtin_amdgcn_wmma_f32_16x16x32_bf16(false, (A), false, (B), (short)0, (C), false, false)

// ---------------------------------------------------------------------------
// CDNA5 async copy: 16B per lane, global -> LDS, tracked by ASYNCcnt.
// ---------------------------------------------------------------------------
__device__ __forceinline__ void async_cp16(void* lds, const void* gaddr) {
  uint32_t l = (uint32_t)(uintptr_t)lds;
  asm volatile("global_load_async_to_lds_b128 %0, %1, off"
               :: "v"(l), "v"(gaddr) : "memory");
}
__device__ __forceinline__ void async_wait0() {
  asm volatile("s_wait_asynccnt 0" ::: "memory");
}

// Load a 16-element bf16 fragment as two 16-byte LDS loads.
__device__ __forceinline__ bf16x16 ld2(const bf16* p0, const bf16* p1) {
  bf16x8 lo = *(const bf16x8*)p0;
  bf16x8 hi = *(const bf16x8*)p1;
  return __builtin_shufflevector(lo, hi, 0,1,2,3,4,5,6,7,8,9,10,11,12,13,14,15);
}

// Pack two bf16 into one dword for b32 stores.
__device__ __forceinline__ uint32_t pack2(bf16 lo, bf16 hi) {
  union { bf16 h[2]; uint32_t u; } p;
  p.h[0] = lo; p.h[1] = hi;
  return p.u;
}

// ---------------------------------------------------------------------------
// One-time precision conversion kernels.
// ---------------------------------------------------------------------------
__global__ __launch_bounds__(256) void cvt_bf16(const float* __restrict__ S,
                                                bf16* __restrict__ D) {
  int i = blockIdx.x * 256 + threadIdx.x;
  f32x4 v = ((const f32x4*)S)[i];
  bf16x4 o;
  o.x = (bf16)v.x; o.y = (bf16)v.y; o.z = (bf16)v.z; o.w = (bf16)v.w;
  ((bf16x4*)D)[i] = o;
}

// S: [K][N] f32 -> D: [N][K] bf16 (transposed), 32x32 tiles via LDS.
__global__ __launch_bounds__(256) void transpose_cvt(const float* __restrict__ S,
                                                     bf16* __restrict__ D,
                                                     int K, int N) {
  __shared__ float t[32][33];
  const int k0 = blockIdx.y * 32, n0 = blockIdx.x * 32;
  const int c = threadIdx.x & 31, r = threadIdx.x >> 5;
  #pragma unroll
  for (int i = 0; i < 4; ++i)
    t[r + i * 8][c] = S[(size_t)(k0 + r + i * 8) * N + n0 + c];
  __syncthreads();
  #pragma unroll
  for (int i = 0; i < 4; ++i)
    D[(size_t)(n0 + r + i * 8) * K + k0 + c] = (bf16)t[c][r + i * 8];
}

// ---------------------------------------------------------------------------
// WMMA GEMM core with double-buffered async staging.
// A[M][K] bf16 row-major, Bt[N][K] bf16 (pre-transposed).
// Block tile 128x128, BK=32; 8 waves = 4(M) x 2(N); wave tile 32x64.
// ---------------------------------------------------------------------------
#define TILE_E (128 * 32)

__device__ __forceinline__ void gemm_stage(const bf16* __restrict__ Ab,
                                           const bf16* __restrict__ Bt,
                                           int K, int mBase, int nBase, int k0,
                                           bf16* As, bf16* Bs) {
  const int tid = threadIdx.x;
  #pragma unroll
  for (int i = 0; i < 2; ++i) {
    int idx = tid + i * 256;         // [0,512)
    int row = idx >> 2, q = idx & 3;
    async_cp16(&As[row * 32 + q * 8], &Ab[(size_t)(mBase + row) * K + k0 + q * 8]);
    async_cp16(&Bs[row * 32 + q * 8], &Bt[(size_t)(nBase + row) * K + k0 + q * 8]);
  }
}

__device__ __forceinline__ void gemm_core(const bf16* __restrict__ Ab,
                                          const bf16* __restrict__ Bt,
                                          int K, int mBase, int nBase,
                                          bf16* AsBuf, bf16* BsBuf, f32x8 acc[2][4]) {
  const int tid  = threadIdx.x;
  const int wave = tid >> 5;
  const int lane = tid & 31;
  const int half = lane >> 4;
  const int l16  = lane & 15;
  const int m0   = (wave >> 1) * 32;
  const int n0   = (wave & 1) * 64;

  gemm_stage(Ab, Bt, K, mBase, nBase, 0, AsBuf, BsBuf);
  async_wait0();
  __syncthreads();

  int buf = 0;
  for (int k0 = 0; k0 < K; k0 += 32) {
    bf16* As = AsBuf + buf * TILE_E;
    bf16* Bs = BsBuf + buf * TILE_E;
    if (k0 + 32 < K)   // prefetch next tile into the other buffer (async)
      gemm_stage(Ab, Bt, K, mBase, nBase, k0 + 32,
                 AsBuf + (buf ^ 1) * TILE_E, BsBuf + (buf ^ 1) * TILE_E);

    const bf16* ar0 = &As[(m0 + l16) * 32 + half * 8];
    const bf16* ar1 = &As[(m0 + 16 + l16) * 32 + half * 8];
    bf16x16 a0 = ld2(ar0, ar0 + 16);
    bf16x16 a1 = ld2(ar1, ar1 + 16);
    #pragma unroll
    for (int ni = 0; ni < 4; ++ni) {
      const bf16* br = &Bs[(n0 + ni * 16 + l16) * 32 + half * 8];
      bf16x16 bfr = ld2(br, br + 16);
      acc[0][ni] = WMMA_BF16(a0, bfr, acc[0][ni]);
      acc[1][ni] = WMMA_BF16(a1, bfr, acc[1][ni]);
    }
    async_wait0();
    __syncthreads();
    buf ^= 1;
  }
}

// ---------------------------------------------------------------------------
// Kernel 1: QKV projection. Q,K -> [B,H,T,D]; V -> [B,H,D,T] (pre-transposed).
// Epilogue: branch + base pointer hoisted per 16-column tile (sel is
// tile-uniform); V stores packed as b32 pairs.
// ---------------------------------------------------------------------------
__global__ __launch_bounds__(256) void gemm_qkv(const bf16* __restrict__ X,
                                                const bf16* __restrict__ Wt,
                                                const float* __restrict__ bias,
                                                bf16* __restrict__ Q,
                                                bf16* __restrict__ Kq,
                                                bf16* __restrict__ V,
                                                int K) {
  __shared__ bf16 As[2 * TILE_E];
  __shared__ bf16 Bs[2 * TILE_E];
  f32x8 acc[2][4] = {};
  const int mBase = blockIdx.y * 128;
  const int nBase = blockIdx.x * 128;
  gemm_core(X, Wt, K, mBase, nBase, As, Bs, acc);

  const int lane = threadIdx.x & 31, wave = threadIdx.x >> 5;
  const int half = lane >> 4, l16 = lane & 15;
  const int m0 = (wave >> 1) * 32, n0 = (wave & 1) * 64;

  const int mb  = mBase + m0;                 // wave row base (= b*T + t)
  const int b   = mb >> 11;                   // T = 2048
  const int tb  = (mb & 2047) + half * 8;     // t base for this lane-half
  const size_t bh16 = (size_t)b * 16;

  #pragma unroll
  for (int ni = 0; ni < 4; ++ni) {
    const int nt  = nBase + n0 + ni * 16;     // tile column base (uniform)
    const int n   = nt + l16;
    const float bia = bias[n];
    const int sel = nt >> 10;                 // 0:q 1:k 2:v (tile-uniform)
    const int c   = n & 1023;
    const int h   = c >> 6, d = c & 63;
    if (sel == 2) {
      // V[bh][d][t]: consecutive r -> consecutive t in the same lane -> b32 pairs
      bf16* vp = V + ((bh16 + h) * 64 + d) * 2048 + tb;
      #pragma unroll
      for (int mi = 0; mi < 2; ++mi)
        #pragma unroll
        for (int rr = 0; rr < 4; ++rr) {
          bf16 lo = (bf16)(acc[mi][ni][2 * rr]     + bia);
          bf16 hi = (bf16)(acc[mi][ni][2 * rr + 1] + bia);
          *(uint32_t*)(vp + mi * 16 + 2 * rr) = pack2(lo, hi);
        }
    } else {
      bf16* qp = (sel ? Kq : Q) + ((bh16 + h) * 2048 + tb) * 64 + d;
      #pragma unroll
      for (int mi = 0; mi < 2; ++mi)
        #pragma unroll
        for (int r = 0; r < 8; ++r)
          qp[(mi * 16 + r) * 64] = (bf16)(acc[mi][ni][r] + bia);
    }
  }
}

// ---------------------------------------------------------------------------
// Kernel 2: causal flash attention. Q,K: [B,H,T,D]; V: [B,H,D,T]; Y: [B,T,C] bf16.
// 128 q-rows per block; double-buffered async K/V staging.
// ---------------------------------------------------------------------------
__global__ __launch_bounds__(256) void flash_attn(const bf16* __restrict__ Q,
                                                  const bf16* __restrict__ Kq,
                                                  const bf16* __restrict__ V,
                                                  bf16* __restrict__ Y) {
  constexpr int T = 2048, D = 64, C = 1024;
  __shared__ bf16 Qs[128 * 64];       // [qrow][d]      16 KB
  __shared__ bf16 Ks[2][32 * 64];     // [key][d]        8 KB
  __shared__ bf16 Vt[2][64 * 32];     // [d][key]        8 KB
  __shared__ bf16 Ps[8][16 * 32];     // per-wave P      8 KB

  const int tid  = threadIdx.x;
  const int wave = tid >> 5;
  const int lane = tid & 31;
  const int half = lane >> 4;
  const int l16  = lane & 15;

  const int bh = blockIdx.y;
  const int b  = bh >> 4;
  const int h  = bh & 15;
  const int q0 = blockIdx.x * 128;

  const bf16* Qg = Q  + (size_t)bh * T * D;   // [t][d]
  const bf16* Kg = Kq + (size_t)bh * T * D;   // [t][d]
  const bf16* Vg = V  + (size_t)bh * D * T;   // [d][t]

  // Stage Q tile 128x64 (4 async b128 per lane) + first K/V tile.
  #pragma unroll
  for (int i = 0; i < 4; ++i) {
    int idx = tid + i * 256;
    int row = idx >> 3, q = idx & 7;
    async_cp16(&Qs[row * 64 + q * 8], &Qg[(size_t)(q0 + row) * 64 + q * 8]);
  }
  {
    int key = tid >> 3, q = tid & 7;
    async_cp16(&Ks[0][key * 64 + q * 8], &Kg[(size_t)key * 64 + q * 8]);
    int dd = tid >> 2, q2 = tid & 3;
    async_cp16(&Vt[0][dd * 32 + q2 * 8], &Vg[(size_t)dd * T + q2 * 8]);
  }
  async_wait0();
  __syncthreads();

  f32x8 O0 = {}, O1 = {}, O2 = {}, O3 = {};
  float mrow[8], lrow[8];
  #pragma unroll
  for (int r = 0; r < 8; ++r) { mrow[r] = -__builtin_inff(); lrow[r] = 0.f; }

  const int qrow_base = q0 + wave * 16 + half * 8;
  const float scale = 0.125f;
  const int kend = q0 + 128;

  int buf = 0;
  for (int kt = 0; kt < kend; kt += 32) {
    if (kt + 32 < kend) {   // async-prefetch next K/V tile into other buffer
      int key = tid >> 3, q = tid & 7;
      async_cp16(&Ks[buf ^ 1][key * 64 + q * 8],
                 &Kg[(size_t)(kt + 32 + key) * 64 + q * 8]);
      int dd = tid >> 2, q2 = tid & 3;
      async_cp16(&Vt[buf ^ 1][dd * 32 + q2 * 8],
                 &Vg[(size_t)dd * T + kt + 32 + q2 * 8]);
    }

    // S = Q * K^T
    const bf16* qrowp = &Qs[(wave * 16 + l16) * 64];
    bf16x16 aQ0 = ld2(qrowp + half * 8,      qrowp + half * 8 + 16);
    bf16x16 aQ1 = ld2(qrowp + 32 + half * 8, qrowp + 32 + half * 8 + 16);
    const bf16* krow0 = &Ks[buf][l16 * 64];
    const bf16* krow1 = &Ks[buf][(16 + l16) * 64];
    bf16x16 bK00 = ld2(krow0 + half * 8,      krow0 + half * 8 + 16);
    bf16x16 bK10 = ld2(krow0 + 32 + half * 8, krow0 + 32 + half * 8 + 16);
    bf16x16 bK01 = ld2(krow1 + half * 8,      krow1 + half * 8 + 16);
    bf16x16 bK11 = ld2(krow1 + 32 + half * 8, krow1 + 32 + half * 8 + 16);

    f32x8 s0 = {}, s1 = {};
    s0 = WMMA_BF16(aQ0, bK00, s0);
    s0 = WMMA_BF16(aQ1, bK10, s0);
    s1 = WMMA_BF16(aQ0, bK01, s1);
    s1 = WMMA_BF16(aQ1, bK11, s1);

    // Online softmax (row = half*8 + r, spread across 16 lanes of one VGPR)
    const int keyb = kt + l16;
    #pragma unroll
    for (int r = 0; r < 8; ++r) {
      int qrow = qrow_base + r;
      float v0 = s0[r] * scale; if (keyb      > qrow) v0 = -__builtin_inff();
      float v1 = s1[r] * scale; if (keyb + 16 > qrow) v1 = -__builtin_inff();
      float tmax = fmaxf(v0, v1);
      tmax = fmaxf(tmax, __shfl_xor(tmax, 1, 16));
      tmax = fmaxf(tmax, __shfl_xor(tmax, 2, 16));
      tmax = fmaxf(tmax, __shfl_xor(tmax, 4, 16));
      tmax = fmaxf(tmax, __shfl_xor(tmax, 8, 16));
      float mnew  = fmaxf(mrow[r], tmax);
      float alpha = __expf(mrow[r] - mnew);
      float p0 = __expf(v0 - mnew);
      float p1 = __expf(v1 - mnew);
      float ps = p0 + p1;
      ps += __shfl_xor(ps, 1, 16);
      ps += __shfl_xor(ps, 2, 16);
      ps += __shfl_xor(ps, 4, 16);
      ps += __shfl_xor(ps, 8, 16);
      lrow[r] = lrow[r] * alpha + ps;
      mrow[r] = mnew;
      O0[r] *= alpha; O1[r] *= alpha; O2[r] *= alpha; O3[r] *= alpha;
      int prow = half * 8 + r;
      Ps[wave][prow * 32 + l16]      = (bf16)p0;
      Ps[wave][prow * 32 + 16 + l16] = (bf16)p1;
    }

    // O += P * V
    const bf16* prowp = &Ps[wave][l16 * 32];
    bf16x16 aP = ld2(prowp + half * 8, prowp + half * 8 + 16);
    const bf16* v0p = &Vt[buf][(l16)      * 32 + half * 16];
    const bf16* v1p = &Vt[buf][(16 + l16) * 32 + half * 16];
    const bf16* v2p = &Vt[buf][(32 + l16) * 32 + half * 16];
    const bf16* v3p = &Vt[buf][(48 + l16) * 32 + half * 16];
    bf16x16 bV0 = ld2(v0p - half * 8, v0p - half * 8 + 16);
    bf16x16 bV1 = ld2(v1p - half * 8, v1p - half * 8 + 16);
    bf16x16 bV2 = ld2(v2p - half * 8, v2p - half * 8 + 16);
    bf16x16 bV3 = ld2(v3p - half * 8, v3p - half * 8 + 16);
    O0 = WMMA_BF16(aP, bV0, O0);
    O1 = WMMA_BF16(aP, bV1, O1);
    O2 = WMMA_BF16(aP, bV2, O2);
    O3 = WMMA_BF16(aP, bV3, O3);

    async_wait0();
    __syncthreads();
    buf ^= 1;
  }

  // Normalize; write y[b][t][h*64+d] bf16 (feeds the bf16 proj GEMM)
  #pragma unroll
  for (int r = 0; r < 8; ++r) {
    float inv = 1.f / lrow[r];
    int t = qrow_base + r;
    bf16* yp = Y + ((size_t)(b * T + t)) * C + h * 64;
    yp[l16]      = (bf16)(O0[r] * inv);
    yp[16 + l16] = (bf16)(O1[r] * inv);
    yp[32 + l16] = (bf16)(O2[r] * inv);
    yp[48 + l16] = (bf16)(O3[r] * inv);
  }
}

// ---------------------------------------------------------------------------
// Kernel 3: output projection (A bf16 [M][K], Bt bf16 [N][K]) -> f32 + bias.
// ---------------------------------------------------------------------------
__global__ __launch_bounds__(256) void gemm_bias(const bf16* __restrict__ A,
                                                 const bf16* __restrict__ Bt,
                                                 const float* __restrict__ bias,
                                                 float* __restrict__ Cg,
                                                 int N, int K) {
  __shared__ bf16 As[2 * TILE_E];
  __shared__ bf16 Bs[2 * TILE_E];
  f32x8 acc[2][4] = {};
  const int mBase = blockIdx.y * 128;
  const int nBase = blockIdx.x * 128;
  gemm_core(A, Bt, K, mBase, nBase, As, Bs, acc);

  const int lane = threadIdx.x & 31, wave = threadIdx.x >> 5;
  const int half = lane >> 4, l16 = lane & 15;
  const int m0 = (wave >> 1) * 32, n0 = (wave & 1) * 64;
  const int mb = mBase + m0 + half * 8;
  #pragma unroll
  for (int ni = 0; ni < 4; ++ni) {
    const int n = nBase + n0 + ni * 16 + l16;
    const float bia = bias[n];
    float* cp = Cg + (size_t)mb * N + n;
    #pragma unroll
    for (int mi = 0; mi < 2; ++mi)
      #pragma unroll
      for (int r = 0; r < 8; ++r)
        cp[(size_t)(mi * 16 + r) * N] = acc[mi][ni][r] + bia;
  }
}

// ---------------------------------------------------------------------------
extern "C" void kernel_launch(void* const* d_in, const int* in_sizes, int n_in,
                              void* d_out, int out_size, void* d_ws, size_t ws_size,
                              hipStream_t stream) {
  (void)in_sizes; (void)n_in; (void)out_size; (void)ws_size;
  constexpr int B = 2, T = 2048, C = 1024, H = 16, D = 64;
  constexpr int M = B * T;            // 4096

  const float* x      = (const float*)d_in[0];
  const float* W_attn = (const float*)d_in[1];
  const float* b_attn = (const float*)d_in[2];
  const float* W_proj = (const float*)d_in[3];
  const float* b_proj = (const float*)d_in[4];
  float* out = (float*)d_out;

  // Workspace (bf16):
  //   xb [M][C] | Wta [3C][C] | Wtp [C][C] | Q,K [B,H,T,D] | V [B,H,D,T] | Y [M][C]
  bf16* xb  = (bf16*)d_ws;
  bf16* Wta = xb  + (size_t)M * C;
  bf16* Wtp = Wta + (size_t)3 * C * C;
  bf16* Qb  = Wtp + (size_t)C * C;
  const size_t nQ = (size_t)B * H * T * D;
  bf16* Kb  = Qb + nQ;
  bf16* Vb  = Kb + nQ;
  bf16* Yb  = Vb + nQ;

  dim3 blk(256);
  cvt_bf16<<<dim3((M * C) / 1024), blk, 0, stream>>>(x, xb);
  transpose_cvt<<<dim3(3 * C / 32, C / 32), blk, 0, stream>>>(W_attn, Wta, C, 3 * C);
  transpose_cvt<<<dim3(C / 32, C / 32), blk, 0, stream>>>(W_proj, Wtp, C, C);
  gemm_qkv<<<dim3(3 * C / 128, M / 128), blk, 0, stream>>>(xb, Wta, b_attn, Qb, Kb, Vb, C);
  flash_attn<<<dim3(T / 128, B * H), blk, 0, stream>>>(Qb, Kb, Vb, Yb);
  gemm_bias<<<dim3(C / 128, M / 128), blk, 0, stream>>>(Yb, Wtp, b_proj, out, C, C);
}
